// GRU_7696581394568
// MI455X (gfx1250) — matrix operand order repaired and checked by reading, not compiled, not run on hardware
//
#include <hip/hip_runtime.h>
#include <hip/hip_bf16.h>

// ---------------------------------------------------------------------------
// CDNA5 (gfx1250) GRU: bf16 WMMA GEMMs (async-to-LDS double-buffered) +
// persistent LDS-resident recurrence with grid-wide barrier.
// ---------------------------------------------------------------------------

typedef __bf16 bf16_t;
typedef __attribute__((ext_vector_type(16))) __bf16        v16bf;
typedef __attribute__((ext_vector_type(8)))  float         v8f;
typedef __attribute__((ext_vector_type(8)))  unsigned int  v8u;
typedef __attribute__((ext_vector_type(4)))  unsigned int  v4u;

constexpr int L_ = 2, B_ = 64, S_ = 512, I_ = 1024, H_ = 1024, O_ = 1024;
constexpr int G3H   = 3 * H_;        // 3072 gate columns
constexpr int MROWS = B_ * S_;       // 32768 (b,s) rows

// ---------------- gfx1250 async global->LDS copy (guarded) -----------------

#if __has_builtin(__builtin_amdgcn_global_load_async_to_lds_b128)
#define HAVE_ASYNC_LDS 1
#else
#define HAVE_ASYNC_LDS 0
#endif

// Builtin prototype (from hipcc diagnostic): first arg is a *global*
// (addrspace(1)) pointer to a 16-byte int vector; LDS side is addrspace(3).
typedef int v4i_vs __attribute__((vector_size(16)));
typedef __attribute__((address_space(1))) v4i_vs* gv4i_p;
typedef __attribute__((address_space(3))) v4i_vs* lv4i_p;

__device__ __forceinline__ void async_copy_b128(bf16_t* lds_dst,
                                                const bf16_t* gsrc) {
#if HAVE_ASYNC_LDS
  __builtin_amdgcn_global_load_async_to_lds_b128((gv4i_p)gsrc, (lv4i_p)lds_dst,
                                                 0, 0);
#else
  *(v4u*)lds_dst = *(const v4u*)gsrc;
#endif
}

__device__ __forceinline__ void async_join() {
#if HAVE_ASYNC_LDS
#if __has_builtin(__builtin_amdgcn_s_wait_asynccnt)
  __builtin_amdgcn_s_wait_asynccnt(0);
#else
  asm volatile("s_wait_asynccnt 0x0" ::: "memory");
#endif
#endif
}

// ---------------- WMMA fragment helpers (per CDNA5 ISA 7.12.2) -------------

__device__ __forceinline__ v8f wmma_bf16(v16bf a, v16bf b, v8f c) {
  // (neg_a, A, neg_b, B, c_mod, C, reuse_a, reuse_b)
  return __builtin_amdgcn_wmma_f32_16x16x32_bf16(false, a, false, b,
                                                 (short)0, c, false, false);
}

// A-matrix 16x32 bf16: lane holds row m=lane&15; lanes<16 K in {0..7,16..23},
// lanes>=16 K in {8..15,24..31}.  Per lane that is two contiguous 16B chunks
// at byte offsets 0 and 32 from (m, 8*hi) -> exactly two b128 loads.
__device__ __forceinline__ v16bf load_a_frag(const bf16_t* __restrict__ base,
                                             int row_stride) {
  const unsigned lane = threadIdx.x & 31u;
  const unsigned m = lane & 15u, hi = lane >> 4;
  const bf16_t* r = base + (size_t)m * row_stride + 8u * hi;
  const v4u lo = *(const v4u*)(r);        // K = base+0..7   (16B)
  const v4u hv = *(const v4u*)(r + 16);   // K = base+16..23 (16B)
  v8u u;
  u[0] = lo[0]; u[1] = lo[1]; u[2] = lo[2]; u[3] = lo[3];
  u[4] = hv[0]; u[5] = hv[1]; u[6] = hv[2]; u[7] = hv[3];
  return __builtin_bit_cast(v16bf, u);
}

// B-matrix 32x16 bf16 fed from row-major W[N][K]: lane holds column
// n=lane&15, K = 8*(lane>>4)+{0..7,16..23} -> two 16B chunks (2x b128).
__device__ __forceinline__ v16bf load_b_frag(const bf16_t* __restrict__ base,
                                             int row_stride) {
  const unsigned lane = threadIdx.x & 31u;
  const unsigned n = lane & 15u, hi = lane >> 4;
  const bf16_t* r = base + (size_t)n * row_stride + 8u * hi;
  const v4u lo = *(const v4u*)(r);
  const v4u hv = *(const v4u*)(r + 16);
  v8u u;
  u[0] = lo[0]; u[1] = lo[1]; u[2] = lo[2]; u[3] = lo[3];
  u[4] = hv[0]; u[5] = hv[1]; u[6] = hv[2]; u[7] = hv[3];
  return __builtin_bit_cast(v16bf, u);
}

// ---------------- generic big GEMM: C[M,N] = A[M,K] * W[N,K]^T + bias ------
// 128x128 workgroup tile, KW=64 K-window, double-buffered async LDS staging,
// 8 waves of 32x64.

constexpr int KW = 64;
constexpr int TS = KW + 8;               // LDS row stride (bf16): 144B
constexpr size_t TILE_ELEMS = 128 * TS;  // one A or B tile
constexpr size_t GEMM_LDS = 4 * TILE_ELEMS * sizeof(bf16_t);  // 2 bufs x (A+B)

__device__ __forceinline__ void stage_tile(const bf16_t* __restrict__ A,
                                           const bf16_t* __restrict__ W,
                                           bf16_t* As, bf16_t* Bs,
                                           int m0, int n0, int K, int k0,
                                           int tid) {
  const int row = tid >> 1, half = tid & 1;  // 32 bf16 per (row,half)
  const bf16_t* ga = A + (size_t)(m0 + row) * K + k0 + half * 32;
  const bf16_t* gw = W + (size_t)(n0 + row) * K + k0 + half * 32;
  bf16_t* la = As + row * TS + half * 32;
  bf16_t* lb = Bs + row * TS + half * 32;
#pragma unroll
  for (int c = 0; c < 4; ++c) {
    async_copy_b128(la + c * 8, ga + c * 8);
    async_copy_b128(lb + c * 8, gw + c * 8);
  }
}

__global__ __launch_bounds__(256) void wmma_gemm_bias(
    const bf16_t* __restrict__ A, const bf16_t* __restrict__ W,
    const float* __restrict__ bias, float* __restrict__ C,
    int M, int N, int K) {
  extern __shared__ char smem[];
  bf16_t* lds = (bf16_t*)smem;   // [buf][A|B][128*TS]

  const int n0   = blockIdx.x * 128;
  const int m0   = blockIdx.y * 128;
  const int tid  = threadIdx.x;
  const int wave = tid >> 5;
  const int wm   = (wave & 3) * 32;   // wave M offset in tile
  const int wn   = (wave >> 2) * 64;  // wave N offset in tile

  const v8f zacc = {};
  v8f acc[2][4];
#pragma unroll
  for (int i = 0; i < 2; ++i)
#pragma unroll
    for (int j = 0; j < 4; ++j) acc[i][j] = zacc;

  const int nk = K / KW;
  stage_tile(A, W, lds, lds + TILE_ELEMS, m0, n0, K, 0, tid);

  for (int i = 0; i < nk; ++i) {
    async_join();
    __syncthreads();
    const int cur = i & 1;
    if (i + 1 < nk) {
      bf16_t* nb = lds + (size_t)(cur ^ 1) * 2 * TILE_ELEMS;
      stage_tile(A, W, nb, nb + TILE_ELEMS, m0, n0, K, (i + 1) * KW, tid);
    }
    const bf16_t* Asc = lds + (size_t)cur * 2 * TILE_ELEMS;
    const bf16_t* Bsc = Asc + TILE_ELEMS;
#pragma unroll
    for (int kk = 0; kk < KW; kk += 32) {
      v16bf af[2], bfv[4];
#pragma unroll
      for (int ii = 0; ii < 2; ++ii)
        af[ii] = load_a_frag(Asc + (wm + ii * 16) * TS + kk, TS);
#pragma unroll
      for (int j = 0; j < 4; ++j)
        bfv[j] = load_b_frag(Bsc + (wn + j * 16) * TS + kk, TS);
#pragma unroll
      for (int ii = 0; ii < 2; ++ii)
#pragma unroll
        for (int j = 0; j < 4; ++j)
          acc[ii][j] = wmma_bf16(af[ii], bfv[j], acc[ii][j]);
    }
  }

  const unsigned lane = tid & 31u;
  const unsigned hi = lane >> 4, ln = lane & 15u;
#pragma unroll
  for (int i = 0; i < 2; ++i)
#pragma unroll
    for (int j = 0; j < 4; ++j) {
      const int col = n0 + wn + j * 16 + (int)ln;
      const float bs = bias[col];
#pragma unroll
      for (int v = 0; v < 8; ++v) {
        const int row = m0 + wm + i * 16 + v + 8 * (int)hi;
        C[(size_t)row * N + col] = acc[i][j][v] + bs;
      }
    }
}

// ---------------- grid-wide generation barrier (L2 atomics) ----------------

__device__ __forceinline__ void grid_barrier(unsigned* arrive, unsigned* release,
                                             unsigned nwg, unsigned target) {
  __threadfence();
  __syncthreads();
  if (threadIdx.x == 0) {
    const unsigned prev = __hip_atomic_fetch_add(arrive, 1u, __ATOMIC_ACQ_REL,
                                                 __HIP_MEMORY_SCOPE_AGENT);
    if (prev == nwg - 1u) {
      __hip_atomic_store(arrive, 0u, __ATOMIC_RELAXED, __HIP_MEMORY_SCOPE_AGENT);
      __hip_atomic_fetch_add(release, 1u, __ATOMIC_RELEASE,
                             __HIP_MEMORY_SCOPE_AGENT);
    } else {
      while (__hip_atomic_load(release, __ATOMIC_ACQUIRE,
                               __HIP_MEMORY_SCOPE_AGENT) < target) {
        __builtin_amdgcn_s_sleep(2);
      }
    }
  }
  __syncthreads();
}

// ---------------- persistent GRU recurrence --------------------------------
// 32 WGs x 384 threads (12 waves). WG g owns h columns [g*32, g*32+32) and
// keeps the matching 96x1024 bf16 Whh slice resident in LDS (~193KB).

constexpr int TJ  = 32;          // h columns per WG
constexpr int NWG = H_ / TJ;     // 32 workgroups
constexpr int WSS = H_ + 8;      // LDS weight row stride (bf16): 2064B
constexpr int GHS = 100;         // LDS gh row stride (f32)
constexpr size_t REC_LDS = (size_t)96 * WSS * sizeof(bf16_t) +
                           (size_t)B_ * GHS * sizeof(float);

__global__ __launch_bounds__(384) void gru_recurrent_layer(
    const float*  __restrict__ gx,      // [B*S, 3H] f32 (precomputed)
    const bf16_t* __restrict__ Whh_b,   // [3H, H] bf16
    const float*  __restrict__ bhh,     // [3H]
    const float*  __restrict__ h0l,     // [B, H]
    float* __restrict__ hA_f, float* __restrict__ hB_f,
    bf16_t* __restrict__ hA_b, bf16_t* __restrict__ hB_b,
    bf16_t* __restrict__ hseq,          // [B*S, H] bf16 (next-layer input)
    float* __restrict__ hn_out,         // [B, H] final hidden -> d_out
    unsigned* __restrict__ sync) {      // {arrive, release}
  extern __shared__ char smem[];
  bf16_t* wsl = (bf16_t*)smem;                                      // Whh slice
  float*  ghs = (float*)(smem + (size_t)96 * WSS * sizeof(bf16_t)); // gh tile

  const int tid  = threadIdx.x;
  const int j0   = blockIdx.x * TJ;
  const unsigned lane = tid & 31u;
  const int wave = tid >> 5;
  const int mt   = (wave & 3) * 16;   // M tile (batch rows)
  const int np   = (wave >> 2);       // 0..2 -> 32 gate-cols each

  // Stage this WG's Whh slice (rows: gate*1024 + j0 + r) into LDS once,
  // via the CDNA5 async global->LDS path.
  for (int idx = tid; idx < 96 * (H_ / 16); idx += blockDim.x) {
    const int r = idx >> 6;            // 0..95
    const int c = idx & 63;            // 16-elem chunk
    const int grow = (r >> 5) * H_ + j0 + (r & 31);
    const bf16_t* g = Whh_b + (size_t)grow * H_ + c * 16;
    bf16_t* l = wsl + (size_t)r * WSS + c * 16;
    async_copy_b128(l, g);
    async_copy_b128(l + 8, g + 8);
  }
  async_join();

  // Init h state (f32 carry + bf16 mirror) for this WG's column slice.
  for (int idx = tid; idx < B_ * TJ; idx += blockDim.x) {
    const int b = idx >> 5, jj = idx & 31;
    const float h = h0l[(size_t)b * H_ + j0 + jj];
    hA_f[(size_t)b * H_ + j0 + jj] = h;
    hA_b[(size_t)b * H_ + j0 + jj] = (bf16_t)h;
  }

  unsigned phase = 0;
  grid_barrier(sync, sync + 1, NWG, ++phase);

  for (int t = 0; t < S_; ++t) {
    const float*  hin_f  = (t & 1) ? hB_f : hA_f;
    const bf16_t* hin_b  = (t & 1) ? hB_b : hA_b;
    float*        hout_f = (t & 1) ? hA_f : hB_f;
    bf16_t*       hout_b = (t & 1) ? hA_b : hB_b;

    // gh[mt:mt+16, np*32 : np*32+32] = h * Whh_sliceT  (K = 1024)
    v8f acc0 = {}, acc1 = {};
    const bf16_t* wb0 = wsl + (size_t)(np * 32)      * WSS;
    const bf16_t* wb1 = wsl + (size_t)(np * 32 + 16) * WSS;
#pragma unroll 4
    for (int k = 0; k < H_; k += 32) {
      const v16bf af = load_a_frag(hin_b + (size_t)mt * H_ + k, H_);
      const v16bf b0 = load_b_frag(wb0 + k, WSS);
      const v16bf b1 = load_b_frag(wb1 + k, WSS);
      acc0 = wmma_bf16(af, b0, acc0);
      acc1 = wmma_bf16(af, b1, acc1);
    }

    const unsigned hi = lane >> 4, ln = lane & 15u;
#pragma unroll
    for (int v = 0; v < 8; ++v) {
      const int m = mt + v + 8 * (int)hi;
      ghs[m * GHS + np * 32 + (int)ln]      = acc0[v];
      ghs[m * GHS + np * 32 + 16 + (int)ln] = acc1[v];
    }
    __syncthreads();

    // Fused gate nonlinearity + state update for the 64x32 slice.
    for (int idx = tid; idx < B_ * TJ; idx += blockDim.x) {
      const int b = idx >> 5, jj = idx & 31;
      const int j = j0 + jj;
      const size_t grow = ((size_t)b * S_ + t) * (size_t)G3H;
      const float hr = ghs[b * GHS + jj]      + bhh[j];
      const float hz = ghs[b * GHS + 32 + jj] + bhh[H_ + j];
      const float hg = ghs[b * GHS + 64 + jj] + bhh[2 * H_ + j];
      const float xr = gx[grow + j];
      const float xz = gx[grow + H_ + j];
      const float xn = gx[grow + 2 * H_ + j];
      // Prefetch next timestep's gx rows (hide HBM behind GEMM + barrier).
      if (t + 1 < S_) {
        __builtin_prefetch(gx + grow + G3H + j, 0, 1);
        __builtin_prefetch(gx + grow + G3H + H_ + j, 0, 1);
        __builtin_prefetch(gx + grow + G3H + 2 * H_ + j, 0, 1);
      }
      const float r  = 1.0f / (1.0f + __expf(-(xr + hr)));
      const float z  = 1.0f / (1.0f + __expf(-(xz + hz)));
      const float n  = tanhf(xn + r * hg);
      const float hp = hin_f[(size_t)b * H_ + j];
      const float hnew = (1.0f - z) * n + z * hp;
      hout_f[(size_t)b * H_ + j] = hnew;
      hout_b[(size_t)b * H_ + j] = (bf16_t)hnew;
      hseq[((size_t)b * S_ + t) * H_ + j] = (bf16_t)hnew;
      if (t == S_ - 1) hn_out[(size_t)b * H_ + j] = hnew;
    }
    grid_barrier(sync, sync + 1, NWG, ++phase);
  }
}

// ---------------- small utility kernels ------------------------------------

__global__ void cvt_f32_to_bf16(const float* __restrict__ src,
                                bf16_t* __restrict__ dst, size_t n) {
  size_t i = (size_t)blockIdx.x * blockDim.x + threadIdx.x;
  const size_t stride = (size_t)gridDim.x * blockDim.x;
  for (; i < n; i += stride) dst[i] = (bf16_t)src[i];
}

__global__ void init_sync(unsigned* s) {
  if (threadIdx.x < 8) s[threadIdx.x] = 0u;
}

// ---------------- host orchestration ---------------------------------------

extern "C" void kernel_launch(void* const* d_in, const int* in_sizes, int n_in,
                              void* d_out, int out_size, void* d_ws, size_t ws_size,
                              hipStream_t stream) {
  (void)in_sizes; (void)n_in; (void)out_size; (void)ws_size;
  const float* x   = (const float*)d_in[0];
  const float* h0  = (const float*)d_in[1];
  const float* Wih = (const float*)d_in[2];
  const float* Whh = (const float*)d_in[3];
  const float* bih = (const float*)d_in[4];
  const float* bhh = (const float*)d_in[5];
  const float* fcW = (const float*)d_in[6];
  const float* fcb = (const float*)d_in[7];
  float* out = (float*)d_out;

  char* ws = (char*)d_ws;
  size_t off = 0;
  auto alloc = [&](size_t bytes) -> void* {
    void* p = ws + off;
    off = (off + bytes + 255) & ~(size_t)255;
    return p;
  };

  bf16_t* xb    = (bf16_t*)alloc((size_t)B_ * S_ * I_ * 2);
  bf16_t* wihb  = (bf16_t*)alloc((size_t)L_ * G3H * I_ * 2);
  bf16_t* whhb  = (bf16_t*)alloc((size_t)L_ * G3H * H_ * 2);
  bf16_t* fcwb  = (bf16_t*)alloc((size_t)O_ * H_ * 2);
  float*  gx    = (float*) alloc((size_t)MROWS * G3H * 4);
  bf16_t* hseqA = (bf16_t*)alloc((size_t)MROWS * H_ * 2);
  bf16_t* hseqB = (bf16_t*)alloc((size_t)MROWS * H_ * 2);
  float*  hAf   = (float*) alloc((size_t)B_ * H_ * 4);
  float*  hBf   = (float*) alloc((size_t)B_ * H_ * 4);
  bf16_t* hAb   = (bf16_t*)alloc((size_t)B_ * H_ * 2);
  bf16_t* hBb   = (bf16_t*)alloc((size_t)B_ * H_ * 2);
  unsigned* sy  = (unsigned*)alloc(256);

  cvt_f32_to_bf16<<<4096, 256, 0, stream>>>(x,   xb,   (size_t)B_ * S_ * I_);
  cvt_f32_to_bf16<<<4096, 256, 0, stream>>>(Wih, wihb, (size_t)L_ * G3H * I_);
  cvt_f32_to_bf16<<<4096, 256, 0, stream>>>(Whh, whhb, (size_t)L_ * G3H * H_);
  cvt_f32_to_bf16<<<1024, 256, 0, stream>>>(fcW, fcwb, (size_t)O_ * H_);
  init_sync<<<1, 32, 0, stream>>>(sy);

  const size_t BSO = (size_t)B_ * S_ * O_;
  dim3 gGate(G3H / 128, MROWS / 128);   // 24 x 256
  dim3 gFc(O_ / 128, MROWS / 128);      // 8 x 256

  // Layer 0
  wmma_gemm_bias<<<gGate, 256, GEMM_LDS, stream>>>(xb, wihb, bih, gx,
                                                   MROWS, G3H, I_);
  gru_recurrent_layer<<<NWG, 384, REC_LDS, stream>>>(
      gx, whhb, bhh, h0, hAf, hBf, hAb, hBb, hseqA, out + BSO, sy);

  // Layer 1
  wmma_gemm_bias<<<gGate, 256, GEMM_LDS, stream>>>(
      hseqA, wihb + (size_t)G3H * I_, bih + G3H, gx, MROWS, G3H, H_);
  gru_recurrent_layer<<<NWG, 384, REC_LDS, stream>>>(
      gx, whhb + (size_t)G3H * H_, bhh + G3H, h0 + (size_t)B_ * H_,
      hAf, hBf, hAb, hBb, hseqB, out + BSO + (size_t)B_ * H_, sy + 2);

  // Final FC: outputs[B,S,O]
  wmma_gemm_bias<<<gFc, 256, GEMM_LDS, stream>>>(hseqB, fcwb, fcb, out,
                                                 MROWS, O_, H_);
}
